// LSTM_44908178047006
// MI455X (gfx1250) — compile-verified, matrix operand order checked by reading, not run on hardware
//
#include <hip/hip_runtime.h>
#include <hip/hip_bf16.h>

// ---------------------------------------------------------------------------
// 2-layer LSTM seq2seq on gfx1250 via v_wmma_f32_16x16x32_bf16.
// Time-major rows: r = t*64 + b,  M = 512*64 = 32768.
// GEMM operands pre-packed into wave32 WMMA fragment layout (one contiguous
// 32B load per lane per 16x32 tile fragment).  Inner loops are register
// double-buffered so loads overlap WMMA issue.
// ---------------------------------------------------------------------------

typedef __attribute__((ext_vector_type(16))) __bf16 v16bf;
typedef __attribute__((ext_vector_type(8)))  float  v8f;

#define WMMA_BF16(a, b, c) \
  __builtin_amdgcn_wmma_f32_16x16x32_bf16(false, (a), false, (b), (short)0, (c), false, false)

// Packed fragment layout (ISA 7.12.2, 16-bit A 16x32):
//   tile (rt, kt), lane l = (r&15) + 16*hi, 16 bf16 per lane (j = 0..15)
//   hi=0: K = {0..7, 16..23}; hi=1: K = {8..15, 24..31}   (within tile)
__device__ __forceinline__ size_t packA_idx(int r, int k, int KT) {
  int kt  = k >> 5;
  int k32 = k & 31;
  int hi  = (k32 >> 3) & 1;
  int j   = (k32 & 7) + ((k32 & 16) ? 8 : 0);
  return ((size_t)(r >> 4) * KT + kt) * 512 + (size_t)(((r & 15) + (hi << 4)) << 4) + j;
}

// ---------------------------------------------------------------------------
__global__ void k_zero32(uint32_t* __restrict__ p, int n) {
  int i = blockIdx.x * 256 + threadIdx.x;
  if (i < n) p[i] = 0u;
}

__global__ void k_bias_sum(const float* __restrict__ a, const float* __restrict__ b,
                           float* __restrict__ o, int n) {
  int i = blockIdx.x * 256 + threadIdx.x;
  if (i < n) o[i] = a[i] + b[i];
}

// Pack W [N,K] row-major f32 into packed B fragments of B = W^T ([K,N]),
// tile order: nt-major, kt contiguous.  out has N*K bf16 elements.
__global__ void k_pack_wT(const float* __restrict__ W, __hip_bfloat16* __restrict__ out,
                          int N, int K) {
  int idx = blockIdx.x * 256 + threadIdx.x;
  if (idx >= N * K) return;
  int within = idx & 511;
  int tile   = idx >> 9;
  int KT = K >> 5;
  int kt = tile % KT;
  int nt = tile / KT;
  int l = within >> 4, j = within & 15;
  int hi = l >> 4;
  int n  = nt * 16 + (l & 15);
  int k  = kt * 32 + ((j < 8) ? (hi * 8 + j) : (16 + hi * 8 + (j - 8)));
  out[idx] = __float2bfloat16(W[(size_t)n * K + k]);
}

// Pack xs = concat(source[b,t,:], target[b,t,:]) as packed-A, rows r=t*64+b, K=320.
__global__ void k_pack_xs(const float* __restrict__ src, const float* __restrict__ tgt,
                          __hip_bfloat16* __restrict__ out) {
  int idx = blockIdx.x * 256 + threadIdx.x;
  const int total = 32768 * 320;
  if (idx >= total) return;
  int within = idx & 511;
  int tile   = idx >> 9;
  const int KT = 10;                     // 320 / 32
  int kt = tile % KT;
  int mt = tile / KT;
  int l = within >> 4, j = within & 15;
  int hi = l >> 4;
  int r  = mt * 16 + (l & 15);
  int k  = kt * 32 + ((j < 8) ? (hi * 8 + j) : (16 + hi * 8 + (j - 8)));
  int b = r & 63, t = r >> 6;
  float v = (k < 256) ? src[((size_t)b * 512 + t) * 256 + k]
                      : tgt[((size_t)b * 513 + t) * 64 + (k - 256)];
  out[idx] = __float2bfloat16(v);
}

// ---------------------------------------------------------------------------
// GEMM: C[M,N] = Apk[M,K] * Bpk[K,N] + bias[n].
// block = 128 (4 waves); each wave: 2 m-tiles x 4 n-tiles (8 accumulators),
// register double-buffered (12 b128 loads per 8 wmma, overlapped).
// grid = (N/64, M/128).
// EPI 0: f32 row-major store (stride N).  EPI 1: ReLU, store packed-A bf16.
template <int EPI>
__global__ void k_gemm(const __hip_bfloat16* __restrict__ Apk,
                       const __hip_bfloat16* __restrict__ Bpk,
                       const float* __restrict__ bias,
                       float* __restrict__ outF,
                       __hip_bfloat16* __restrict__ outPk,
                       int KT, int N) {
  int wave = threadIdx.x >> 5;
  int lane = threadIdx.x & 31;
  int mt0  = (blockIdx.y * 4 + wave) * 2;
  int nt0  = blockIdx.x * 4;

  const v16bf* A0 = (const v16bf*)Apk + (size_t)(mt0 + 0) * KT * 32 + lane;
  const v16bf* A1 = (const v16bf*)Apk + (size_t)(mt0 + 1) * KT * 32 + lane;
  const v16bf* B0 = (const v16bf*)Bpk + (size_t)(nt0 + 0) * KT * 32 + lane;
  const v16bf* B1 = (const v16bf*)Bpk + (size_t)(nt0 + 1) * KT * 32 + lane;
  const v16bf* B2 = (const v16bf*)Bpk + (size_t)(nt0 + 2) * KT * 32 + lane;
  const v16bf* B3 = (const v16bf*)Bpk + (size_t)(nt0 + 3) * KT * 32 + lane;

  v8f c00 = {}, c01 = {}, c02 = {}, c03 = {};
  v8f c10 = {}, c11 = {}, c12 = {}, c13 = {};

  // Prologue loads (iteration 0 fragments).
  v16bf a0 = A0[0], a1 = A1[0];
  v16bf b0 = B0[0], b1 = B1[0], b2 = B2[0], b3 = B3[0];

  for (int kt = 0; kt + 1 < KT; ++kt) {
    size_t o = (size_t)(kt + 1) * 32;
    // Next-iteration fragments issue before this iteration's WMMAs.
    v16bf a0n = A0[o], a1n = A1[o];
    v16bf b0n = B0[o], b1n = B1[o], b2n = B2[o], b3n = B3[o];
    if (kt + 2 < KT) {  // stream-ahead on the activation operand
      __builtin_prefetch((const void*)(A0 + (size_t)(kt + 2) * 32), 0, 0);
      __builtin_prefetch((const void*)(A1 + (size_t)(kt + 2) * 32), 0, 0);
    }
    c00 = WMMA_BF16(a0, b0, c00);
    c01 = WMMA_BF16(a0, b1, c01);
    c02 = WMMA_BF16(a0, b2, c02);
    c03 = WMMA_BF16(a0, b3, c03);
    c10 = WMMA_BF16(a1, b0, c10);
    c11 = WMMA_BF16(a1, b1, c11);
    c12 = WMMA_BF16(a1, b2, c12);
    c13 = WMMA_BF16(a1, b3, c13);
    a0 = a0n; a1 = a1n;
    b0 = b0n; b1 = b1n; b2 = b2n; b3 = b3n;
  }
  c00 = WMMA_BF16(a0, b0, c00);
  c01 = WMMA_BF16(a0, b1, c01);
  c02 = WMMA_BF16(a0, b2, c02);
  c03 = WMMA_BF16(a0, b3, c03);
  c10 = WMMA_BF16(a1, b0, c10);
  c11 = WMMA_BF16(a1, b1, c11);
  c12 = WMMA_BF16(a1, b2, c12);
  c13 = WMMA_BF16(a1, b3, c13);

  int nl = lane & 15;
  int mh = (lane >> 4) * 8;
  v8f acc[2][4] = {{c00, c01, c02, c03}, {c10, c11, c12, c13}};
#pragma unroll
  for (int mi = 0; mi < 2; ++mi) {
#pragma unroll
    for (int q = 0; q < 4; ++q) {
      int n = (nt0 + q) * 16 + nl;
      float bb = bias[n];
#pragma unroll
      for (int i = 0; i < 8; ++i) {
        int m = (mt0 + mi) * 16 + mh + i;
        float v = acc[mi][q][i] + bb;
        if (EPI == 0) {
          outF[(size_t)m * N + n] = v;
        } else {
          v = v > 0.f ? v : 0.f;
          outPk[packA_idx(m, n, N >> 5)] = __float2bfloat16(v);
        }
      }
    }
  }
}

// ---------------------------------------------------------------------------
// One LSTM time step:  gates = xp_t + h_prev @ W_hh^T, then pointwise update.
// grid = 64 blocks (one per 16 hidden units), block = 128 (4 waves = all 64
// batch rows).  Gate columns {n, n+1024, n+2048, n+3072} are the four n-tiles
// q*64+hb, so this block finishes its hidden slice end-to-end.  The xp/cell
// scalars are loaded up-front so their latency hides under the WMMA loop, and
// the W_hh fragment stream is register double-buffered.
__global__ void k_lstm_step(const __hip_bfloat16* __restrict__ hprev,  // packed A [64 x 1024]
                            const __hip_bfloat16* __restrict__ Whh,    // packed B [1024 x 4096]
                            const float* __restrict__ xp_t,            // [64 x 4096]
                            float* __restrict__ c,                     // [64 x 1024]
                            __hip_bfloat16* __restrict__ hout) {       // packed A [64 x 1024]
  int wave = threadIdx.x >> 5;
  int lane = threadIdx.x & 31;
  int hb   = blockIdx.x;  // hidden tile 0..63

  int nl = lane & 15;
  int mh = (lane >> 4) * 8;
  int nc = hb * 16 + nl;

  // Hoist epilogue operands: gate pre-activations from xp and cell state.
  float xi[8], xf[8], xg[8], xo[8], cold[8];
#pragma unroll
  for (int i = 0; i < 8; ++i) {
    int b = wave * 16 + mh + i;
    const float* xr = xp_t + (size_t)b * 4096;
    xi[i]   = xr[nc];
    xf[i]   = xr[1024 + nc];
    xg[i]   = xr[2048 + nc];
    xo[i]   = xr[3072 + nc];
    cold[i] = c[(size_t)b * 1024 + nc];
  }

  const v16bf* A  = (const v16bf*)hprev + (size_t)wave * 32 * 32 + lane;  // KT = 32
  const v16bf* B0 = (const v16bf*)Whh + (size_t)(0 * 64 + hb) * 32 * 32 + lane;
  const v16bf* B1 = (const v16bf*)Whh + (size_t)(1 * 64 + hb) * 32 * 32 + lane;
  const v16bf* B2 = (const v16bf*)Whh + (size_t)(2 * 64 + hb) * 32 * 32 + lane;
  const v16bf* B3 = (const v16bf*)Whh + (size_t)(3 * 64 + hb) * 32 * 32 + lane;

  v8f g0 = {}, g1 = {}, g2 = {}, g3 = {};
  v16bf a = A[0];
  v16bf b0 = B0[0], b1 = B1[0], b2 = B2[0], b3 = B3[0];
  for (int kt = 0; kt + 1 < 32; ++kt) {
    size_t o = (size_t)(kt + 1) * 32;
    v16bf an  = A[o];
    v16bf b0n = B0[o], b1n = B1[o], b2n = B2[o], b3n = B3[o];
    g0 = WMMA_BF16(a, b0, g0);
    g1 = WMMA_BF16(a, b1, g1);
    g2 = WMMA_BF16(a, b2, g2);
    g3 = WMMA_BF16(a, b3, g3);
    a = an; b0 = b0n; b1 = b1n; b2 = b2n; b3 = b3n;
  }
  g0 = WMMA_BF16(a, b0, g0);
  g1 = WMMA_BF16(a, b1, g1);
  g2 = WMMA_BF16(a, b2, g2);
  g3 = WMMA_BF16(a, b3, g3);

#pragma unroll
  for (int i = 0; i < 8; ++i) {
    int b = wave * 16 + mh + i;
    float gi = g0[i] + xi[i];
    float gf = g1[i] + xf[i];
    float gg = g2[i] + xg[i];
    float go = g3[i] + xo[i];
    float si = 1.f / (1.f + __expf(-gi));
    float sf = 1.f / (1.f + __expf(-gf));
    float so = 1.f / (1.f + __expf(-go));
    float tg = tanhf(gg);
    float cn = sf * cold[i] + si * tg;
    float hn = so * tanhf(cn);
    c[(size_t)b * 1024 + nc] = cn;
    hout[packA_idx(b, nc, 32)] = __float2bfloat16(hn);
  }
}

// ---------------------------------------------------------------------------
extern "C" void kernel_launch(void* const* d_in, const int* in_sizes, int n_in,
                              void* d_out, int out_size, void* d_ws, size_t ws_size,
                              hipStream_t stream) {
  const float* source = (const float*)d_in[0];
  const float* target = (const float*)d_in[1];
  const float* W_ih0  = (const float*)d_in[2];
  const float* W_hh0  = (const float*)d_in[3];
  const float* b_ih0  = (const float*)d_in[4];
  const float* b_hh0  = (const float*)d_in[5];
  const float* W_ih1  = (const float*)d_in[6];
  const float* W_hh1  = (const float*)d_in[7];
  const float* b_ih1  = (const float*)d_in[8];
  const float* b_hh1  = (const float*)d_in[9];
  const float* W1     = (const float*)d_in[10];
  const float* b1     = (const float*)d_in[11];
  const float* W2     = (const float*)d_in[12];
  const float* b2     = (const float*)d_in[13];
  const float* W3     = (const float*)d_in[14];
  const float* b3     = (const float*)d_in[15];

  char*  ws = (char*)d_ws;
  size_t off = 0;
  auto alloc = [&](size_t bytes) -> char* {
    char* p = ws + off;
    off = (off + bytes + 255) & ~(size_t)255;
    return p;
  };

  const size_t M = 32768;        // 512 * 64 (time-major rows)
  const size_t STEP = 64 * 1024; // bf16 elems of one packed h step-slot

  float*          xp     = (float*)alloc(M * 4096 * 4);            // 512 MB (reused L0/L1)
  __hip_bfloat16* xs_pk  = (__hip_bfloat16*)alloc(M * 320 * 2);
  __hip_bfloat16* hseq0  = (__hip_bfloat16*)alloc(513 * STEP * 2); // slot 0 = h(-1) = 0
  __hip_bfloat16* hseq1  = (__hip_bfloat16*)alloc(513 * STEP * 2);
  __hip_bfloat16* y1pk   = (__hip_bfloat16*)alloc(M * 512 * 2);
  __hip_bfloat16* y2pk   = (__hip_bfloat16*)alloc(M * 128 * 2);
  __hip_bfloat16* Wih0pk = (__hip_bfloat16*)alloc(4096 * 320 * 2);
  __hip_bfloat16* Whh0pk = (__hip_bfloat16*)alloc(4096 * 1024 * 2);
  __hip_bfloat16* Wih1pk = (__hip_bfloat16*)alloc(4096 * 1024 * 2);
  __hip_bfloat16* Whh1pk = (__hip_bfloat16*)alloc(4096 * 1024 * 2);
  __hip_bfloat16* W1pk   = (__hip_bfloat16*)alloc(512 * 1024 * 2);
  __hip_bfloat16* W2pk   = (__hip_bfloat16*)alloc(128 * 512 * 2);
  __hip_bfloat16* W3pk   = (__hip_bfloat16*)alloc(64 * 128 * 2);
  float*          bsum0  = (float*)alloc(4096 * 4);
  float*          bsum1  = (float*)alloc(4096 * 4);
  float*          c0     = (float*)alloc(64 * 1024 * 4);
  float*          c1     = (float*)alloc(64 * 1024 * 4);
  (void)ws_size; (void)in_sizes; (void)n_in; (void)out_size;

  auto zero = [&](void* p, size_t words) {
    k_zero32<<<dim3((unsigned)((words + 255) / 256)), 256, 0, stream>>>((uint32_t*)p, (int)words);
  };
  // h(-1) slots, cell states, and output row 0.
  zero(hseq0, STEP / 2);
  zero(hseq1, STEP / 2);
  zero(c0, 64 * 1024);
  zero(c1, 64 * 1024);
  zero(d_out, 64 * 64);

  auto packw = [&](const float* W, __hip_bfloat16* out, int N, int K) {
    int total = N * K;
    k_pack_wT<<<dim3((total + 255) / 256), 256, 0, stream>>>(W, out, N, K);
  };
  packw(W_ih0, Wih0pk, 4096, 320);
  packw(W_hh0, Whh0pk, 4096, 1024);
  packw(W_ih1, Wih1pk, 4096, 1024);
  packw(W_hh1, Whh1pk, 4096, 1024);
  packw(W1, W1pk, 512, 1024);
  packw(W2, W2pk, 128, 512);
  packw(W3, W3pk, 64, 128);

  k_bias_sum<<<dim3(16), 256, 0, stream>>>(b_ih0, b_hh0, bsum0, 4096);
  k_bias_sum<<<dim3(16), 256, 0, stream>>>(b_ih1, b_hh1, bsum1, 4096);

  k_pack_xs<<<dim3((32768 * 320 + 255) / 256), 256, 0, stream>>>(source, target, xs_pk);

  // Layer 0: xp = xs @ W_ih0^T + (b_ih0 + b_hh0)
  k_gemm<0><<<dim3(64, 256), 128, 0, stream>>>(xs_pk, Wih0pk, bsum0, xp, nullptr, 10, 4096);
  for (int t = 0; t < 512; ++t) {
    k_lstm_step<<<dim3(64), 128, 0, stream>>>(hseq0 + (size_t)t * STEP, Whh0pk,
                                              xp + (size_t)t * 64 * 4096, c0,
                                              hseq0 + (size_t)(t + 1) * STEP);
  }

  // Layer 1: xp = h0 @ W_ih1^T + (b_ih1 + b_hh1)
  k_gemm<0><<<dim3(64, 256), 128, 0, stream>>>(hseq0 + STEP, Wih1pk, bsum1, xp, nullptr, 32, 4096);
  for (int t = 0; t < 512; ++t) {
    k_lstm_step<<<dim3(64), 128, 0, stream>>>(hseq1 + (size_t)t * STEP, Whh1pk,
                                              xp + (size_t)t * 64 * 4096, c1,
                                              hseq1 + (size_t)(t + 1) * STEP);
  }

  // Dense head: relu(h1@W1^T+b1) -> relu(.@W2^T+b2) -> .@W3^T+b3
  k_gemm<1><<<dim3(8, 256), 128, 0, stream>>>(hseq1 + STEP, W1pk, b1, nullptr, y1pk, 32, 512);
  k_gemm<1><<<dim3(2, 256), 128, 0, stream>>>(y1pk, W2pk, b2, nullptr, y2pk, 16, 128);
  // Final: rows r = t*64+b land at d_out[(t+1)*64*64 + b*64 + n] (row 0 stays zero).
  k_gemm<0><<<dim3(1, 256), 128, 0, stream>>>(y2pk, W3pk, b3, (float*)d_out + 4096, nullptr, 4, 64);
}
